// Sam_Block_14645838479457
// MI455X (gfx1250) — matrix operand order repaired and checked by reading, not compiled, not run on hardware
//
#include <hip/hip_runtime.h>

typedef __bf16  v16bf __attribute__((ext_vector_type(16)));
typedef __bf16  v8bfv __attribute__((ext_vector_type(8)));
typedef float   v8f   __attribute__((ext_vector_type(8)));
typedef unsigned int u32x4 __attribute__((ext_vector_type(4)));
typedef int     i32x4 __attribute__((ext_vector_type(4)));
typedef int     i32x8 __attribute__((ext_vector_type(8)));

#define SCALE_C 0.125f     // 64^-0.5
#define NEG_INF (-1e30f)

// ---- CDNA5 data-movement feature detection (compile-safe on both toolchains) ----
#if __has_builtin(__builtin_amdgcn_tensor_load_to_lds) && __has_builtin(__builtin_amdgcn_s_wait_tensorcnt)
#define HAVE_TDM 1
#endif
#if __has_builtin(__builtin_amdgcn_global_load_async_to_lds_b128) && __has_builtin(__builtin_amdgcn_s_wait_asynccnt)
#define HAVE_ASYNC 1
#endif

#define AS1P(p) ((__attribute__((address_space(1))) void*)(unsigned long long)(uintptr_t)(p))
#define AS3P(p) ((__attribute__((address_space(3))) void*)(unsigned)(uintptr_t)(p))

// ---------------- WMMA helpers (CDNA5 gfx1250, wave32) ----------------

__device__ inline v8f wmma_bf16(v16bf a, v16bf b, v8f c) {
  // D = A(16x32 bf16) * B(32x16 bf16) + C(16x16 f32)
  return __builtin_amdgcn_wmma_f32_16x16x32_bf16(false, a, false, b, (short)0, c, false, false);
}

// A-matrix 16x32 fragment, row-major source with 'stride' elements between rows.
// Layout (ISA 7.12.2): lane m=lane&15, half=lane>>4;
//   elems 0..7  -> K = 8*half + 0..7 ; elems 8..15 -> K = 16 + 8*half + 0..7
__device__ inline v16bf frag_a(const __bf16* base, int stride) {
  const int lane = threadIdx.x & 31;
  const int m = lane & 15, half = lane >> 4;
  const __bf16* r = base + m * stride + 8 * half;
  v8bfv lo = *(const v8bfv*)(r);
  v8bfv hi = *(const v8bfv*)(r + 16);
  v16bf o;
#pragma unroll
  for (int i = 0; i < 8; ++i) { o[i] = lo[i]; o[i + 8] = hi[i]; }
  return o;
}

// B-matrix 32x16 fragment from K-contiguous (transposed) storage Bt[n][k]:
//   lane n=lane&15, half=lane>>4; elems 0..7 -> K = 8*half + 0..7 ; elems 8..15 -> K = 16 + 8*half + 0..7
__device__ inline v16bf frag_b(const __bf16* base, int stride) {
  const int lane = threadIdx.x & 31;
  const int n = lane & 15, half = lane >> 4;
  const __bf16* r = base + n * stride + 8 * half;
  v8bfv lo = *(const v8bfv*)(r);
  v8bfv hi = *(const v8bfv*)(r + 16);
  v16bf o;
#pragma unroll
  for (int i = 0; i < 8; ++i) { o[i] = lo[i]; o[i + 8] = hi[i]; }
  return o;
}

#ifdef HAVE_TDM
// Issue (no wait) a TDM 2D-tile load: A[m0:m0+128, kb:kb+32] bf16 -> LDS buffer
// (row-major [128][32]). OOB rows/cols return zero via tensor_dim clamps.
__device__ inline void tdm_load_a_tile(const __bf16* A, int M, int K,
                                       int m0, int kb, __bf16* lbuf)
{
  unsigned long long ga = (unsigned long long)(uintptr_t)(A + (size_t)m0 * K + kb);
  unsigned lds = (unsigned)(uintptr_t)(void*)lbuf;
  unsigned dim0 = (unsigned)(K - kb);        // remaining elems in a row (OOB -> 0)
  unsigned dim1 = (unsigned)(M - m0);        // remaining rows (OOB rows -> 0)
  unsigned long long st0 = (unsigned long long)(unsigned)K;  // row stride (elems)
  u32x4 g0;
  g0[0] = 1u;                                          // count=1, user descriptor
  g0[1] = lds;                                         // lds_addr (bytes)
  g0[2] = (unsigned)ga;                                // global_addr[31:0]
  g0[3] = (unsigned)(ga >> 32) | 0x80000000u;          // global_addr[56:32] | type=2
  i32x8 g1;
  g1[0] = (int)(1u << 16);                             // wg_mask=0, data_size=1 (2B)
  g1[1] = (int)((dim0 & 0xFFFFu) << 16);               // tensor_dim0[15:0] @ bit48
  g1[2] = (int)((dim0 >> 16) | ((dim1 & 0xFFFFu) << 16)); // dim0 hi | dim1 lo
  g1[3] = (int)((dim1 >> 16) | (32u << 16));           // dim1 hi | tile_dim0=32
  g1[4] = (int)(128u);                                 // tile_dim1=128, tile_dim2=0
  g1[5] = (int)(unsigned)st0;                          // tensor_dim0_stride[31:0]
  g1[6] = (int)((st0 >> 32) & 0xFFFFu);                // stride0[47:32] | dim1_stride lo=0
  g1[7] = 0;
  i32x4 g2; g2[0] = 0; g2[1] = 0; g2[2] = 0; g2[3] = 0;
  i32x4 g3; g3[0] = 0; g3[1] = 0; g3[2] = 0; g3[3] = 0;
#if __clang_major__ >= 23
  i32x8 g4;
#pragma unroll
  for (int z = 0; z < 8; ++z) g4[z] = 0;
  __builtin_amdgcn_tensor_load_to_lds(g0, g1, g2, g3, g4, 0);
#else
  __builtin_amdgcn_tensor_load_to_lds(g0, g1, g2, g3, 0);
#endif
}
#endif

// ---------------- Generic LDS-staged bf16 GEMM ----------------
// C[M,N] = act(A[M,K] @ B[K,N] + bias) (+ residual), out bf16 or f32.
// Block tile 128x128, K-step 32, 8 waves of 32x64 each. N%128==0, K%32==0.
// A tile: double-buffered TDM DMA (TENSORcnt) when available; else async-to-LDS; else sync.

__global__ __launch_bounds__(256) void gemm_bf16(
    const __bf16* __restrict__ A, const __bf16* __restrict__ B,
    const float* __restrict__ bias, float* __restrict__ Cf,
    __bf16* __restrict__ Cb, const float* __restrict__ residual,
    int M, int N, int K, int act)
{
  __shared__ __bf16 As[2][128 * 32];
  __shared__ __bf16 Bs[128 * 32];   // transposed: Bs[n][k]
  const int tid = threadIdx.x;
  const int nbn = N >> 7;
  const int bn = blockIdx.x % nbn, bm = blockIdx.x / nbn;
  const int m0 = bm << 7, n0 = bn << 7;
  const int wid = tid >> 5, lane = tid & 31;
  const int wm = (wid & 3) << 5;    // 0,32,64,96
  const int wn = (wid >> 2) << 6;   // 0,64

  v8f acc[2][4];
#pragma unroll
  for (int i = 0; i < 2; ++i)
#pragma unroll
    for (int j = 0; j < 4; ++j)
#pragma unroll
      for (int r = 0; r < 8; ++r) acc[i][j][r] = 0.f;

  const int nk = K >> 5;
#ifdef HAVE_TDM
  if (tid < 32) tdm_load_a_tile(A, M, K, m0, 0, As[0]);   // prologue DMA
#endif

  for (int ik = 0; ik < nk; ++ik) {
    const int kb = ik << 5;
    const int buf = ik & 1;
#if defined(HAVE_TDM)
    // issue next tile's DMA into the other buffer; overlap with B staging below
    if (tid < 32 && (ik + 1) < nk)
      tdm_load_a_tile(A, M, K, m0, kb + 32, As[buf ^ 1]);
#elif defined(HAVE_ASYNC)
    // ---- async copy global -> LDS (ASYNCcnt), 16B units ----
#pragma unroll
    for (int u = 0; u < 2; ++u) {
      int i = tid + u * 256;                 // 0..511
      int row = i >> 2, seg = i & 3;
      int gr = m0 + row;
      __bf16* lp = As[buf] + row * 32 + seg * 8;
      if (gr < M) {
        __builtin_amdgcn_global_load_async_to_lds_b128(
            AS1P(A + (size_t)gr * K + kb + seg * 8), AS3P(lp), 0, 0);
      } else {
        v8bfv vz;
#pragma unroll
        for (int e = 0; e < 8; ++e) vz[e] = (__bf16)0.f;
        *(v8bfv*)lp = vz;
      }
    }
#else
    // ---- synchronous staging fallback ----
#pragma unroll
    for (int u = 0; u < 2; ++u) {
      int i = tid + u * 256;                 // 0..511
      int row = i >> 2, seg = i & 3;
      int gr = m0 + row;
      v8bfv vv;
      if (gr < M) vv = *(const v8bfv*)(A + (size_t)gr * K + kb + seg * 8);
      else {
#pragma unroll
        for (int e = 0; e < 8; ++e) vv[e] = (__bf16)0.f;
      }
      *(v8bfv*)(As[buf] + row * 32 + seg * 8) = vv;
    }
    if (kb + 32 < K) {
      int gr = m0 + (tid >> 1);
      if (gr < M) __builtin_prefetch(A + (size_t)gr * K + kb + 32 + (tid & 1) * 16, 0, 1);
    }
#endif
    // stage B transposed: read [k][n] 16B, scatter to Bs[n][k]
#pragma unroll
    for (int u = 0; u < 2; ++u) {
      int i = tid + u * 256;
      int k = i >> 4, nseg = i & 15;
      v8bfv vv = *(const v8bfv*)(B + (size_t)(kb + k) * N + n0 + nseg * 8);
#pragma unroll
      for (int e = 0; e < 8; ++e) Bs[(nseg * 8 + e) * 32 + k] = vv[e];
    }
#if defined(HAVE_TDM)
    // current buffer's DMA is the older outstanding op; TDM completes in order
    if (tid < 32) {
      if ((ik + 1) < nk) __builtin_amdgcn_s_wait_tensorcnt(1);
      else               __builtin_amdgcn_s_wait_tensorcnt(0);
    }
#elif defined(HAVE_ASYNC)
    __builtin_amdgcn_s_wait_asynccnt(0);
#endif
    __syncthreads();

    v16bf af[2], bfr[4];
#pragma unroll
    for (int i = 0; i < 2; ++i) af[i] = frag_a(As[buf] + (wm + i * 16) * 32, 32);
#pragma unroll
    for (int j = 0; j < 4; ++j) bfr[j] = frag_b(Bs + (wn + j * 16) * 32, 32);
#pragma unroll
    for (int i = 0; i < 2; ++i)
#pragma unroll
      for (int j = 0; j < 4; ++j)
        acc[i][j] = wmma_bf16(af[i], bfr[j], acc[i][j]);
    __syncthreads();
  }

  const int half = lane >> 4, ln = lane & 15;
#pragma unroll
  for (int i = 0; i < 2; ++i) {
#pragma unroll
    for (int j = 0; j < 4; ++j) {
      int col = n0 + wn + j * 16 + ln;
      float bv = bias ? bias[col] : 0.f;
#pragma unroll
      for (int r = 0; r < 8; ++r) {
        int row = m0 + wm + i * 16 + r + 8 * half;
        if (row < M) {
          float v = acc[i][j][r] + bv;
          if (act) v = 0.5f * v * (1.f + erff(v * 0.70710678118f));   // exact GELU
          if (residual) v += residual[(size_t)row * N + col];
          if (Cb) Cb[(size_t)row * N + col] = (__bf16)v;
          else    Cf[(size_t)row * N + col] = v;
        }
      }
    }
  }
}

// ---------------- LayerNorm helpers ----------------

__device__ inline float block_reduce_sum(float v, float* red) {
#pragma unroll
  for (int off = 16; off; off >>= 1) v += __shfl_xor(v, off, 32);
  int wid = threadIdx.x >> 5, lane = threadIdx.x & 31;
  __syncthreads();
  if (lane == 0) red[wid] = v;
  __syncthreads();
  float s = (threadIdx.x < 8) ? red[threadIdx.x] : 0.f;
#pragma unroll
  for (int off = 4; off; off >>= 1) s += __shfl_xor(s, off, 32);
  if (threadIdx.x == 0) red[0] = s;
  __syncthreads();
  return red[0];
}

// LN1 + window partition (64x64 -> pad 70x70 -> 50 windows x 196 tokens, padded to 224 rows)
__global__ __launch_bounds__(256) void ln1_window(
    const float* __restrict__ x, const float* __restrict__ g,
    const float* __restrict__ b, __bf16* __restrict__ xw)
{
  __shared__ float red[8];
  int row = blockIdx.x;                 // 0..11199
  int w = row / 224, t = row % 224;
  int bimg = w / 25, wi = w % 25;
  int ph = (wi / 5) * 14 + t / 14, pw = (wi % 5) * 14 + t % 14;
  __bf16* dst = xw + (size_t)row * 768;
  bool valid = (t < 196) && (ph < 64) && (pw < 64);
  if (!valid) {
    for (int i = threadIdx.x; i < 768; i += 256) dst[i] = (__bf16)0.f;
    return;
  }
  const float* src = x + (((size_t)bimg * 64 + ph) * 64 + pw) * 768;
  float s = 0.f;
  for (int i = threadIdx.x; i < 768; i += 256) s += src[i];
  float mean = block_reduce_sum(s, red) * (1.f / 768.f);
  float vs = 0.f;
  for (int i = threadIdx.x; i < 768; i += 256) { float d = src[i] - mean; vs += d * d; }
  float var = block_reduce_sum(vs, red) * (1.f / 768.f);
  float inv = rsqrtf(var + 1e-5f);
  for (int i = threadIdx.x; i < 768; i += 256)
    dst[i] = (__bf16)((src[i] - mean) * inv * g[i] + b[i]);
}

// LN over straight rows (x2 -> bf16)
__global__ __launch_bounds__(256) void ln_rows(
    const float* __restrict__ x, const float* __restrict__ g,
    const float* __restrict__ b, __bf16* __restrict__ o)
{
  __shared__ float red[8];
  int row = blockIdx.x;
  const float* src = x + (size_t)row * 768;
  __bf16* dst = o + (size_t)row * 768;
  float s = 0.f;
  for (int i = threadIdx.x; i < 768; i += 256) s += src[i];
  float mean = block_reduce_sum(s, red) * (1.f / 768.f);
  float vs = 0.f;
  for (int i = threadIdx.x; i < 768; i += 256) { float d = src[i] - mean; vs += d * d; }
  float var = block_reduce_sum(vs, red) * (1.f / 768.f);
  float inv = rsqrtf(var + 1e-5f);
  for (int i = threadIdx.x; i < 768; i += 256)
    dst[i] = (__bf16)((src[i] - mean) * inv * g[i] + b[i]);
}

// ---------------- rel-pos bias tables ----------------
// rh[w,h,q,kh] = dot(q_row, rel_pos_h[qh-kh+13]);  rw analogous with qw-kw.
__global__ void rel_tab(const __bf16* __restrict__ qkv,
                        const float* __restrict__ rph, const float* __restrict__ rpw,
                        float* __restrict__ rh, float* __restrict__ rw)
{
  int idx = blockIdx.x * 256 + threadIdx.x;
  if (idx >= 50 * 12 * 196 * 14) return;
  int i = idx % 14;
  int t = (idx / 14) % 196;
  int h = (idx / (14 * 196)) % 12;
  int w = idx / (14 * 196 * 12);
  const __bf16* q = qkv + ((size_t)(w * 224 + t)) * 2304 + h * 64;
  int qh = t / 14, qw = t % 14;
  const float* Rh = rph + (qh - i + 13) * 64;
  const float* Rw = rpw + (qw - i + 13) * 64;
  float sh = 0.f, sw = 0.f;
#pragma unroll 8
  for (int d = 0; d < 64; ++d) {
    float qv = (float)q[d];
    sh += qv * Rh[d];
    sw += qv * Rw[d];
  }
  rh[idx] = sh;
  rw[idx] = sw;
}

// ---------------- Flash-style windowed attention ----------------
// grid: 50 windows * 12 heads * 2 halves; 224 threads (7 waves), wave = one 16-row q tile.
__global__ __launch_bounds__(224) void attn_win(
    const __bf16* __restrict__ qkv, const float* __restrict__ rh,
    const float* __restrict__ rw, __bf16* __restrict__ outp)
{
  __shared__ __bf16 vt[64 * 224];     // V transposed: vt[d][j]
  __shared__ __bf16 pst[7][16 * 32];  // per-wave P tile staging
  const int tid = threadIdx.x;
  const int bi = blockIdx.x;
  const int w = bi / 24;
  const int h = (bi % 24) >> 1;
  const int part = bi & 1;

  // stage V^T cooperatively: one token row per thread
  {
    const __bf16* vb = qkv + ((size_t)w * 224) * 2304 + 1536 + h * 64;
    int j = tid;
#pragma unroll
    for (int ds = 0; ds < 8; ++ds) {
      v8bfv vv = *(const v8bfv*)(vb + (size_t)j * 2304 + ds * 8);
#pragma unroll
      for (int e = 0; e < 8; ++e) vt[(ds * 8 + e) * 224 + j] = vv[e];
    }
  }
  __syncthreads();

  const int wid = tid >> 5, lane = tid & 31;
  const int half = lane >> 4, ln = lane & 15;
  const int m0 = (part * 7 + wid) * 16;

  const __bf16* qb = qkv + ((size_t)(w * 224 + m0)) * 2304 + h * 64;
  v16bf aq0 = frag_a(qb, 2304);        // d 0..31
  v16bf aq1 = frag_a(qb + 32, 2304);   // d 32..63

  v8f o[4];
#pragma unroll
  for (int t = 0; t < 4; ++t)
#pragma unroll
    for (int r = 0; r < 8; ++r) o[t][r] = 0.f;
  float mr[8], ls[8];
#pragma unroll
  for (int r = 0; r < 8; ++r) { mr[r] = -3.0e38f; ls[r] = 0.f; }

  const float* rhb = rh + ((size_t)(w * 12 + h)) * 196 * 14;
  const float* rwb = rw + ((size_t)(w * 12 + h)) * 196 * 14;

  for (int jb = 0; jb < 7; ++jb) {
    int j0 = jb * 32;
    v8f s[2];
#pragma unroll
    for (int sub = 0; sub < 2; ++sub) {
#pragma unroll
      for (int r = 0; r < 8; ++r) s[sub][r] = 0.f;
      int n0 = j0 + sub * 16;
      const __bf16* kb = qkv + ((size_t)(w * 224 + n0)) * 2304 + 768 + h * 64;
      v16bf b0 = frag_b(kb, 2304);       // contraction d 0..31 (contiguous in k rows)
      v16bf b1 = frag_b(kb + 32, 2304);  // d 32..63
      s[sub] = wmma_bf16(aq0, b0, s[sub]);
      s[sub] = wmma_bf16(aq1, b1, s[sub]);
      int col = n0 + ln;
      bool cv = col < 196;
      int kh = cv ? col / 14 : 0, kw = cv ? col % 14 : 0;
#pragma unroll
      for (int r = 0; r < 8; ++r) {
        if (cv) {
          int m = m0 + r + 8 * half;
          int mc = (m < 196) ? m : 0;
          s[sub][r] = s[sub][r] * SCALE_C + rhb[mc * 14 + kh] + rwb[mc * 14 + kw];
        } else {
          s[sub][r] = NEG_INF;
        }
      }
    }
    // online softmax; each 16-lane group (same half) holds one row per accumulator elem
#pragma unroll
    for (int r = 0; r < 8; ++r) {
      float mx = fmaxf(s[0][r], s[1][r]);
#pragma unroll
      for (int off = 8; off; off >>= 1) mx = fmaxf(mx, __shfl_xor(mx, off, 32));
      float mn = fmaxf(mr[r], mx);
      float corr = __expf(mr[r] - mn);
      float p0 = __expf(s[0][r] - mn);
      float p1 = __expf(s[1][r] - mn);
      float rs = p0 + p1;
#pragma unroll
      for (int off = 8; off; off >>= 1) rs += __shfl_xor(rs, off, 32);
      ls[r] = ls[r] * corr + rs;
      mr[r] = mn;
#pragma unroll
      for (int t = 0; t < 4; ++t) o[t][r] *= corr;
      pst[wid][(r + 8 * half) * 32 + ln]      = (__bf16)p0;
      pst[wid][(r + 8 * half) * 32 + 16 + ln] = (__bf16)p1;
    }
    // O += P(16x32) @ V(32x64) via WMMA; B-fragments read from vt (K contiguous)
    v16bf pa = frag_a(&pst[wid][0], 32);
#pragma unroll
    for (int t = 0; t < 4; ++t) {
      v16bf vb = frag_b(&vt[(t * 16) * 224 + j0], 224);
      o[t] = wmma_bf16(pa, vb, o[t]);
    }
  }

#pragma unroll
  for (int r = 0; r < 8; ++r) {
    float inv = 1.f / ls[r];
    int m = m0 + r + 8 * half;
    size_t ro = ((size_t)(w * 224 + m)) * 768 + h * 64;
#pragma unroll
    for (int t = 0; t < 4; ++t) outp[ro + t * 16 + ln] = (__bf16)(o[t][r] * inv);
  }
}

// ---------------- misc elementwise ----------------

__global__ void cvt_bf16(const float* __restrict__ s, __bf16* __restrict__ d, int n) {
  int i = blockIdx.x * 256 + threadIdx.x;
  if (i < n) d[i] = (__bf16)s[i];
}

__global__ void unwindow_add(const float* __restrict__ x,
                             const float* __restrict__ pf,
                             float* __restrict__ x2)
{
  int idx = blockIdx.x * 256 + threadIdx.x;
  if (idx >= 8192 * 768) return;
  int c = idx % 768, tok = idx / 768;
  int bimg = tok >> 12, hp = (tok >> 6) & 63, wp = tok & 63;
  int win = bimg * 25 + (hp / 14) * 5 + (wp / 14);
  int t = (hp % 14) * 14 + (wp % 14);
  x2[idx] = x[idx] + pf[((size_t)(win * 224 + t)) * 768 + c];
}

// ---------------- launch ----------------

extern "C" void kernel_launch(void* const* d_in, const int* in_sizes, int n_in,
                              void* d_out, int out_size, void* d_ws, size_t ws_size,
                              hipStream_t stream)
{
  const float* x     = (const float*)d_in[0];
  const float* ln1g  = (const float*)d_in[1];
  const float* ln1b  = (const float*)d_in[2];
  const float* qkvw  = (const float*)d_in[3];
  const float* qkvb  = (const float*)d_in[4];
  const float* projw = (const float*)d_in[5];
  const float* projb = (const float*)d_in[6];
  const float* rph   = (const float*)d_in[7];
  const float* rpw   = (const float*)d_in[8];
  const float* ln2g  = (const float*)d_in[9];
  const float* ln2b  = (const float*)d_in[10];
  const float* w1    = (const float*)d_in[11];
  const float* b1    = (const float*)d_in[12];
  const float* w2    = (const float*)d_in[13];
  const float* b2    = (const float*)d_in[14];
  (void)in_sizes; (void)n_in; (void)out_size; (void)ws_size;

  char* p = (char*)d_ws;
  auto carve = [&](size_t bytes) -> char* {
    char* r = p; p += (bytes + 255) & ~(size_t)255; return r;
  };
  __bf16* qkvw_b = (__bf16*)carve(768ull * 2304 * 2);
  __bf16* projw_b= (__bf16*)carve(768ull * 768 * 2);
  __bf16* w1_b   = (__bf16*)carve(768ull * 3072 * 2);
  __bf16* w2_b   = (__bf16*)carve(3072ull * 768 * 2);
  __bf16* xw     = (__bf16*)carve(11200ull * 768 * 2);
  __bf16* qkv    = (__bf16*)carve(11200ull * 2304 * 2);
  float*  rh     = (float*)carve(50ull * 12 * 196 * 14 * 4);
  float*  rw     = (float*)carve(50ull * 12 * 196 * 14 * 4);
  __bf16* ao     = (__bf16*)carve(11200ull * 768 * 2);
  float*  pf     = (float*)carve(11200ull * 768 * 4);
  float*  x2     = (float*)carve(8192ull * 768 * 4);
  __bf16* xh     = (__bf16*)carve(8192ull * 768 * 2);
  __bf16* h1     = (__bf16*)carve(8192ull * 3072 * 2);

  auto cgrid = [](long n) { return dim3((unsigned)((n + 255) / 256)); };

  cvt_bf16<<<cgrid(768 * 2304), 256, 0, stream>>>(qkvw, qkvw_b, 768 * 2304);
  cvt_bf16<<<cgrid(768 * 768),  256, 0, stream>>>(projw, projw_b, 768 * 768);
  cvt_bf16<<<cgrid(768 * 3072), 256, 0, stream>>>(w1, w1_b, 768 * 3072);
  cvt_bf16<<<cgrid(3072 * 768), 256, 0, stream>>>(w2, w2_b, 3072 * 768);

  ln1_window<<<11200, 256, 0, stream>>>(x, ln1g, ln1b, xw);

  // QKV: [11200,768] x [768,2304] -> bf16
  gemm_bf16<<<dim3(88 * 18), 256, 0, stream>>>(xw, qkvw_b, qkvb, nullptr, qkv, nullptr,
                                               11200, 2304, 768, 0);

  rel_tab<<<cgrid(50 * 12 * 196 * 14), 256, 0, stream>>>(qkv, rph, rpw, rh, rw);

  attn_win<<<1200, 224, 0, stream>>>(qkv, rh, rw, ao);

  // proj: [11200,768] x [768,768] -> f32
  gemm_bf16<<<dim3(88 * 6), 256, 0, stream>>>(ao, projw_b, projb, pf, nullptr, nullptr,
                                              11200, 768, 768, 0);

  unwindow_add<<<cgrid(8192 * 768), 256, 0, stream>>>(x, pf, x2);

  ln_rows<<<8192, 256, 0, stream>>>(x2, ln2g, ln2b, xh);

  // MLP1 + exact GELU: [8192,768] x [768,3072] -> bf16
  gemm_bf16<<<dim3(64 * 24), 256, 0, stream>>>(xh, w1_b, b1, nullptr, h1, nullptr,
                                               8192, 3072, 768, 1);

  // MLP2 + residual(x2): [8192,3072] x [3072,768] -> d_out (f32)
  gemm_bf16<<<dim3(64 * 6), 256, 0, stream>>>(h1, w2_b, b2, (float*)d_out, nullptr, x2,
                                              8192, 768, 3072, 0);
}
